// NNUEModel_7121055777504
// MI455X (gfx1250) — matrix operand order repaired and checked by reading, not compiled, not run on hardware
//
#include <hip/hip_runtime.h>

// NNUE forward pass for MI455X (gfx1250, wave32).
// B=4096, K=32, NF=22528, L1=1024, L2=15, L3=32, COUNT=9.
//
// Roofline: FT gather = 1.0 GiB reads, all L2-resident (W_ft = 88 MiB < 192 MB L2)
// -> L2-bandwidth bound. GEMM l0p@W1 = 1.2 GFLOP -> done with v_wmma_f32_16x16x32_f16
// (f16 in / f32 acc), dense over all 9 buckets (144 = 9 x 16-col tiles), bucket
// selected in the tail kernel.

#define B_    4096
#define K_    32
#define L1_   1024
#define L2_   15
#define L3_   32
#define CNT_  9
#define N1_   (CNT_ * (L2_ + 1))   // 144 = 9 bucket tiles of 16 columns

typedef __attribute__((ext_vector_type(16))) _Float16 v16h;
typedef __attribute__((ext_vector_type(8)))  _Float16 v8h;
typedef __attribute__((ext_vector_type(4)))  _Float16 v4h;
typedef __attribute__((ext_vector_type(8)))  float    v8f;
typedef __attribute__((ext_vector_type(4)))  float    v4f;

__device__ __forceinline__ float clamp01(float x) { return fminf(fmaxf(x, 0.0f), 1.0f); }

// ---------------------------------------------------------------------------
// Kernel 1: feature transform. One workgroup (256 thr) per batch row.
// Thread t accumulates dims [4t, 4t+3] with float4 (b128) gathers from W_ft
// (L2-resident), applies perspective mix + clip, then pairs (j, j+512) via LDS
// and emits the 1024-dim l0p row as f16 for the WMMA GEMM.
// ---------------------------------------------------------------------------
__global__ __launch_bounds__(256)
void ft_kernel(const float* __restrict__ us,  const float* __restrict__ them,
               const int*   __restrict__ wI,  const float* __restrict__ wV,
               const int*   __restrict__ bI,  const float* __restrict__ bV,
               const float* __restrict__ Wft, const float* __restrict__ bft,
               _Float16*    __restrict__ l0p)
{
    __shared__ float sA[L1_];   // clipped l0[:, 0:1024]
    __shared__ float sB[L1_];   // clipped l0[:, 1024:2048]
    const int b  = blockIdx.x;
    const int t  = threadIdx.x;
    const int d0 = t * 4;

    v4f wacc = *(const v4f*)(bft + d0);
    v4f bacc = wacc;
    const int base = b * K_;

    #pragma unroll 4
    for (int k = 0; k < K_; ++k) {
        const int   idx = wI[base + k];
        const float val = wV[base + k];
        const v4f   row = *(const v4f*)(Wft + (size_t)idx * L1_ + d0);
        wacc += val * row;
    }
    #pragma unroll 4
    for (int k = 0; k < K_; ++k) {
        const int   idx = bI[base + k];
        const float val = bV[base + k];
        const v4f   row = *(const v4f*)(Wft + (size_t)idx * L1_ + d0);
        bacc += val * row;
    }

    const float u = us[b], v = them[b];
    #pragma unroll
    for (int j = 0; j < 4; ++j) {
        sA[d0 + j] = clamp01(u * wacc[j] + v * bacc[j]);
        sB[d0 + j] = clamp01(u * bacc[j] + v * wacc[j]);
    }
    __syncthreads();

    v4h outv;
    #pragma unroll
    for (int j = 0; j < 4; ++j) {
        const int   p = d0 + j;
        const float r = (p < 512) ? (sA[p] * sA[p + 512])
                                  : (sB[p - 512] * sB[p]);
        outv[j] = (_Float16)(r * (127.0f / 128.0f));
    }
    *(v4h*)(l0p + (size_t)b * L1_ + d0) = outv;
}

// ---------------------------------------------------------------------------
// Kernel 2: W1 (1024x144 f32, row-major) -> W1T (144x1024 f16, row-major).
// Puts each output column contiguous in K so B-fragment lanes load a single
// contiguous 32-byte run (same contiguous k order as the A fragment).
// ---------------------------------------------------------------------------
__global__ __launch_bounds__(256)
void convW1_kernel(const float* __restrict__ W1, _Float16* __restrict__ W1T)
{
    const int tid = blockIdx.x * 256 + threadIdx.x;
    if (tid >= N1_ * L1_) return;
    const int n = tid / L1_;
    const int k = tid - n * L1_;
    W1T[tid] = (_Float16)W1[k * N1_ + n];
}

// ---------------------------------------------------------------------------
// Kernel 3: GEMM  C(4096x144) = A(4096x1024,f16) * W1T^T  via WMMA.
// 8 waves/block, each wave owns one 16-row M-tile and all 9 N-tiles:
// A fragment loaded once per K-step, reused by 9 v_wmma_f32_16x16x32_f16.
// Fragment layouts used here (contiguous k order, same for A and B):
//   A 16x32 f16 : lane ln=lane&15 row, hf=lane>>4 -> K run [hf*16, +15]
//   B 32x16 f16 : lane holds col n=ln, K run [hf*16, +15] (contiguous in W1T row)
//   D 16x16 f32 : acc[r] -> (M = r + hf*8, N = ln)
// ---------------------------------------------------------------------------
__global__ __launch_bounds__(256)
void gemm_kernel(const _Float16* __restrict__ A,
                 const _Float16* __restrict__ Bt,
                 float*          __restrict__ C)
{
    const int wave = threadIdx.x >> 5;
    const int lane = threadIdx.x & 31;
    const int ln   = lane & 15;
    const int hf   = lane >> 4;
    const int m0   = (blockIdx.x * 8 + wave) * 16;

    v8f acc[CNT_];
    const v8f zero = {};
    #pragma unroll
    for (int n = 0; n < CNT_; ++n) acc[n] = zero;

    const _Float16* Arow = A + (size_t)(m0 + ln) * L1_;

    for (int kb = 0; kb < L1_; kb += 32) {
        union { v16h v; v8h h[2]; } af;
        af.h[0] = *(const v8h*)(Arow + kb + hf * 16);
        af.h[1] = *(const v8h*)(Arow + kb + hf * 16 + 8);
        #pragma unroll
        for (int n = 0; n < CNT_; ++n) {
            const v16h bf = *(const v16h*)(Bt + (size_t)(n * 16 + ln) * L1_ + kb + hf * 16);
            acc[n] = __builtin_amdgcn_wmma_f32_16x16x32_f16(
                false, af.v, false, bf, (short)0, acc[n], false, false);
        }
    }

    #pragma unroll
    for (int n = 0; n < CNT_; ++n) {
        #pragma unroll
        for (int r = 0; r < 8; ++r) {
            C[(size_t)(m0 + r + hf * 8) * N1_ + n * 16 + ln] = acc[n][r];
        }
    }
}

// ---------------------------------------------------------------------------
// Kernel 4: tail. One wave32 per batch row. Bucket-select 16 cols of l1full,
// add b1, split l1x/l1f, build the 30 squared/linear features via lane
// broadcast, matvec against the 30x32 bucket slice of W2 (lane j = col j),
// clip, dot with W3[:,bucket] via shuffle reduction, add b3 + l1f skip.
// ---------------------------------------------------------------------------
__global__ __launch_bounds__(256)
void tail_kernel(const int*   __restrict__ lsi,
                 const float* __restrict__ l1full,
                 const float* __restrict__ b1,
                 const float* __restrict__ W2, const float* __restrict__ b2,
                 const float* __restrict__ W3, const float* __restrict__ b3,
                 float*       __restrict__ out)
{
    const int wave = threadIdx.x >> 5;
    const int lane = threadIdx.x & 31;
    const int b    = blockIdx.x * 8 + wave;
    const int bkt  = lsi[b];

    float c = 0.0f;
    if (lane < 16)
        c = l1full[(size_t)b * N1_ + bkt * 16 + lane] + b1[bkt * 16 + lane];
    const float l1f = __shfl(c, 15, 32);       // pass-through feature (col L2)
    const float l1x = clamp01(c);              // valid in lanes 0..14

    float acc = b2[bkt * L3_ + lane];
    #pragma unroll
    for (int i = 0; i < 2 * L2_; ++i) {
        const float xi = __shfl(l1x, (i < L2_) ? i : (i - L2_), 32);
        const float fi = ((i < L2_) ? xi * xi : xi) * (127.0f / 128.0f);
        acc += fi * W2[i * (CNT_ * L3_) + bkt * L3_ + lane];
    }
    const float l2x = clamp01(acc);

    float p = l2x * W3[lane * CNT_ + bkt];
    #pragma unroll
    for (int off = 16; off > 0; off >>= 1)
        p += __shfl_down(p, off, 32);

    if (lane == 0)
        out[b] = p + b3[bkt] + l1f;
}

// ---------------------------------------------------------------------------
extern "C" void kernel_launch(void* const* d_in, const int* in_sizes, int n_in,
                              void* d_out, int out_size, void* d_ws, size_t ws_size,
                              hipStream_t stream)
{
    const float* us   = (const float*)d_in[0];
    const float* them = (const float*)d_in[1];
    const int*   wI   = (const int*)  d_in[2];
    const float* wV   = (const float*)d_in[3];
    const int*   bI   = (const int*)  d_in[4];
    const float* bV   = (const float*)d_in[5];
    const int*   lsi  = (const int*)  d_in[6];
    const float* Wft  = (const float*)d_in[7];
    const float* bft  = (const float*)d_in[8];
    const float* W1   = (const float*)d_in[9];
    const float* b1   = (const float*)d_in[10];
    const float* W2   = (const float*)d_in[11];
    const float* b2   = (const float*)d_in[12];
    const float* W3   = (const float*)d_in[13];
    const float* b3   = (const float*)d_in[14];
    float* out = (float*)d_out;

    // Workspace layout (~10.6 MiB total):
    //   l0p    : B*L1 f16   = 8 MiB
    //   W1T    : N1*L1 f16  = 288 KiB
    //   l1full : B*N1 f32   = 2.25 MiB
    char* ws = (char*)d_ws;
    _Float16* l0p    = (_Float16*)ws;
    _Float16* W1T    = (_Float16*)(ws + (size_t)B_ * L1_ * sizeof(_Float16));
    float*    l1full = (float*)   (ws + (size_t)B_ * L1_ * sizeof(_Float16)
                                      + (size_t)N1_ * L1_ * sizeof(_Float16));

    ft_kernel<<<B_, 256, 0, stream>>>(us, them, wI, wV, bI, bV, Wft, bft, l0p);
    convW1_kernel<<<(N1_ * L1_ + 255) / 256, 256, 0, stream>>>(W1, W1T);
    gemm_kernel<<<B_ / (16 * 8), 256, 0, stream>>>(l0p, W1T, l1full);
    tail_kernel<<<B_ / 8, 256, 0, stream>>>(lsi, l1full, b1, W2, b2, W3, b3, out);
}
